// EdgeDecoder_28372553957634
// MI455X (gfx1250) — matrix operand order repaired and checked by reading, not compiled, not run on hardware
//
#include <hip/hip_runtime.h>
#include <stdint.h>

typedef __attribute__((ext_vector_type(16))) __bf16 v16bf;
typedef __attribute__((ext_vector_type(8)))  float  v8f;

#define CDIM 128   // per-node embedding / GEMM1 N
#define KDIM 256   // concat width / GEMM1 K

// f32 -> bf16 (round-half-up) packed pair: 2x v_add + 1x v_perm_b32.
// Selector 0x07060302: D = { hi16(y), hi16(x) } (bytes 0-3 from src1=x, 4-7 from src0=y)
__device__ __forceinline__ unsigned pack_bf16x2(float x, float y) {
    unsigned ux = __float_as_uint(x) + 0x8000u;
    unsigned uy = __float_as_uint(y) + 0x8000u;
#if __has_builtin(__builtin_amdgcn_perm)
    return __builtin_amdgcn_perm(uy, ux, 0x07060302u);
#else
    return (ux >> 16) | (uy & 0xffff0000u);
#endif
}

__global__ __launch_bounds__(256)
void edge_decoder_kernel(const float* __restrict__ materials,
                         const float* __restrict__ concepts,
                         const long long* __restrict__ eli,
                         const float* __restrict__ W1,
                         const float* __restrict__ b1,
                         const float* __restrict__ W2,
                         const float* __restrict__ b2,
                         float* __restrict__ outSm,
                         float* __restrict__ outK,
                         int E)
{
    // W1^T as bf16: row n (0..127) holds K=0..255; 32B chunks XOR-swizzled by n
    // to spread LDS banks. Exactly 64 KB.
    __shared__ unsigned short Bsh[CDIM * KDIM];

    const int tid = threadIdx.x;

    // ---- stage W1 (f32 [128n][256k], row-major) into LDS as bf16 ----
    for (int idx = tid; idx < CDIM * (KDIM / 2); idx += 256) {
        int n  = idx >> 7;          // 0..127
        int kp = idx & 127;         // float2 pair index, k = 2*kp
        float2 v = ((const float2*)(W1 + (size_t)n * KDIM))[kp];
        unsigned pk = pack_bf16x2(v.x, v.y);
        int chunk = kp >> 3;                 // 16-element (32B) chunk 0..15
        int swz   = chunk ^ (n & 15);
        *((unsigned*)((char*)Bsh + n * 512 + swz * 32 + (kp & 7) * 4)) = pk;
    }
    __syncthreads();

    const int wave = tid >> 5;
    const int lane = tid & 31;
    const int lm   = lane & 15;    // A: row M within tile; B/C: col N within tile
    const int hf   = lane >> 4;    // half select for K striping
    const int base = blockIdx.x * 128 + wave * 16;

    int e  = base + lm;
    int eC = (e < E) ? e : (E - 1);
    const bool valid = (e < E);

    const long long row = eli[eC];
    const long long col = eli[(size_t)E + eC];
    const float* mrow = materials + (size_t)row * CDIM;
    const float* crow = concepts  + (size_t)col * CDIM;
    float* kout = outK + (size_t)eC * KDIM;

    v8f acc[8] = {};   // C tiles: [16 x 128] f32

    // ---- GEMM1: k[16x256]_bf16 x W1^T[256x128]_bf16 -> f32 ----
    #pragma unroll
    for (int t = 0; t < 8; ++t) {
        // K-range [32t, 32t+32): tiles 0..3 from materials, 4..7 from concepts
        const float* src = (t < 4) ? (mrow + 32 * t) : (crow + 32 * t - CDIM);
        // A layout (16-bit 16x32): lane half hf covers K = hf*8+[0..7] and 16+hf*8+[0..7]
        float4 c1a = *((const float4*)(src + hf * 8));
        float4 c1b = *((const float4*)(src + hf * 8 + 4));
        float4 c2a = *((const float4*)(src + 16 + hf * 8));
        float4 c2b = *((const float4*)(src + 16 + hf * 8 + 4));

        // issue all 8 B-fragment LDS loads up front so waits can be spread
        union BU { v16bf v; uint4 q[2]; };
        BU B[8];
        #pragma unroll
        for (int nt = 0; nt < 8; ++nt) {
            int n     = nt * 16 + lm;
            int chunk0 = (2 * t) ^ (n & 15);
            int chunk1 = (2 * t + 1) ^ (n & 15);
            B[nt].q[0] = *((const uint4*)((const char*)Bsh + n * 512 + chunk0 * 32 + hf * 16));
            B[nt].q[1] = *((const uint4*)((const char*)Bsh + n * 512 + chunk1 * 32 + hf * 16));
        }

        if (valid) {  // these chunks ARE the k output (each element exactly once)
            *((float4*)(kout + 32 * t + hf * 8))          = c1a;
            *((float4*)(kout + 32 * t + hf * 8 + 4))      = c1b;
            *((float4*)(kout + 32 * t + 16 + hf * 8))     = c2a;
            *((float4*)(kout + 32 * t + 16 + hf * 8 + 4)) = c2b;
        }

        union { v16bf v; unsigned u[8]; } A;
        A.u[0] = pack_bf16x2(c1a.x, c1a.y);
        A.u[1] = pack_bf16x2(c1a.z, c1a.w);
        A.u[2] = pack_bf16x2(c1b.x, c1b.y);
        A.u[3] = pack_bf16x2(c1b.z, c1b.w);
        A.u[4] = pack_bf16x2(c2a.x, c2a.y);
        A.u[5] = pack_bf16x2(c2a.z, c2a.w);
        A.u[6] = pack_bf16x2(c2b.x, c2b.y);
        A.u[7] = pack_bf16x2(c2b.z, c2b.w);

        #pragma unroll
        for (int nt = 0; nt < 8; ++nt) {
            acc[nt] = __builtin_amdgcn_wmma_f32_16x16x32_bf16(
                false, A.v, false, B[nt].v, (short)0, acc[nt], false, false);
        }
    }

    // ---- bias + ReLU (C layout: lane -> N, vgpr j -> M = j + 8*hf) ----
    #pragma unroll
    for (int nt = 0; nt < 8; ++nt) {
        float bias = b1[nt * 16 + lm];
        #pragma unroll
        for (int j = 0; j < 8; ++j)
            acc[nt][j] = fmaxf(acc[nt][j] + bias, 0.f);
    }

    // ---- GEMM2 (N=2) on VALU + 16-lane butterfly reduce + softmax ----
    float w2a[8], w2b[8];
    #pragma unroll
    for (int nt = 0; nt < 8; ++nt) {
        w2a[nt] = W2[nt * 16 + lm];
        w2b[nt] = W2[CDIM + nt * 16 + lm];
    }
    const float bias0 = b2[0], bias1 = b2[1];

    #pragma unroll
    for (int j = 0; j < 8; ++j) {
        float s0 = 0.f, s1 = 0.f;
        #pragma unroll
        for (int nt = 0; nt < 8; ++nt) {
            s0 += acc[nt][j] * w2a[nt];
            s1 += acc[nt][j] * w2b[nt];
        }
        #pragma unroll
        for (int m = 1; m < 16; m <<= 1) {
            s0 += __shfl_xor(s0, m, 16);
            s1 += __shfl_xor(s1, m, 16);
        }
        s0 += bias0; s1 += bias1;
        float mx = fmaxf(s0, s1);
        float e0 = __expf(s0 - mx), e1 = __expf(s1 - mx);
        float inv = 1.0f / (e0 + e1);
        if (lm == 0) {
            int mr = j + 8 * hf;        // M index of this vgpr in the C tile
            int ee = base + mr;
            if (ee < E) {
                float2 p; p.x = e0 * inv; p.y = e1 * inv;
                *((float2*)(outSm + 2 * (size_t)ee)) = p;
            }
        }
    }
}

extern "C" void kernel_launch(void* const* d_in, const int* in_sizes, int n_in,
                              void* d_out, int out_size, void* d_ws, size_t ws_size,
                              hipStream_t stream) {
    const float*     materials = (const float*)d_in[0];
    const float*     concepts  = (const float*)d_in[1];
    const long long* eli       = (const long long*)d_in[2];   // int64 [2,E]
    const float*     W1        = (const float*)d_in[3];
    const float*     b1        = (const float*)d_in[4];
    const float*     W2        = (const float*)d_in[5];
    const float*     b2        = (const float*)d_in[6];

    const int E = in_sizes[2] / 2;
    float* outSm = (float*)d_out;                 // softmax [E,2]
    float* outK  = outSm + (size_t)E * 2;         // k [E,256]

    const int grid = (E + 127) / 128;             // 8 waves x 16 edges per block
    edge_decoder_kernel<<<grid, 256, 0, stream>>>(
        materials, concepts, eli, W1, b1, W2, b2, outSm, outK, E);
}
